// Retention_60112362275374
// MI455X (gfx1250) — matrix operand order repaired and checked by reading, not compiled, not run on hardware
//
#include <hip/hip_runtime.h>
#include <math.h>

// Problem constants (match reference)
#define B_  2
#define L_  2048
#define D_  2048
#define H_  16
#define HD_ 128
#define EPS_ 1e-5f
#define SCALE_Q 0.08838834764831845f  // HD^-0.5

typedef __attribute__((ext_vector_type(16))) __bf16 v16bf;
typedef __attribute__((ext_vector_type(8)))  __bf16 v8bf;
typedef __attribute__((ext_vector_type(8)))  float  v8f;

union BF16x16 { v16bf v; v8bf h[2]; };

static __device__ __forceinline__ v8f vzero8() {
  v8f v;
#pragma unroll
  for (int e = 0; e < 8; ++e) v[e] = 0.0f;
  return v;
}

static __device__ __forceinline__ v8f wmma_bf16(v16bf a, v16bf b, v8f c) {
  return __builtin_amdgcn_wmma_f32_16x16x32_bf16(false, a, false, b, (short)0, c,
                                                 false, false);
}

// A fragment (16x32, bf16): lane holds row (m0 + lane%16); K halves at
// c0=(lane>>4)*8 -> elems [c0..c0+7] then [c0+16..c0+23].
static __device__ __forceinline__ v16bf load_afrag(const __bf16* p /*row base*/,
                                                   int c0) {
  BF16x16 u;
  u.h[0] = *(const v8bf*)(p + c0);
  u.h[1] = *(const v8bf*)(p + c0 + 16);
  return u.v;
}
// B fragment (32x16, bf16): lane holds col (n0 + lane%16); K halves at
// k0=(lane>>4)*8 -> elems [k0..k0+7] then [k0+16..k0+23].
static __device__ __forceinline__ v16bf load_bfrag(const __bf16* p /*col base*/,
                                                   int k0) {
  BF16x16 u;
  u.h[0] = *(const v8bf*)(p + k0);
  u.h[1] = *(const v8bf*)(p + k0 + 16);
  return u.v;
}

// ---------------------------------------------------------------------------
// Kernel 1: Y = x @ W^T  (per batch), output bf16 in [B,H,L,HD] layout.
// Tile: 128(M=L) x 128(N=D) per workgroup; N-tile == one head (HD=128).
// K staged 64 at a time through LDS -> 16 WMMA per barrier pair.
// ---------------------------------------------------------------------------
__global__ __launch_bounds__(256) void ret_proj_gemm(
    const float* __restrict__ x, const float* __restrict__ W,
    __bf16* __restrict__ out, float scale) {
  __shared__ __bf16 As[128][72];  // [m][k0..k63], row stride 144B (16B aligned)
  __shared__ __bf16 Bs[128][72];  // [n][k0..k63]

  const int tid  = threadIdx.x;
  const int wid  = tid >> 5;
  const int lane = tid & 31;
  const int l0 = blockIdx.x * 128;
  const int n0 = blockIdx.y * 128;
  const int b  = blockIdx.z;
  const int wm = (wid & 3) * 32;   // 2 m-subtiles: wm, wm+16
  const int wn = (wid >> 2) * 64;  // 4 n-subtiles: wn..wn+48

  v8f acc[2][4];
#pragma unroll
  for (int i = 0; i < 2; ++i)
#pragma unroll
    for (int j = 0; j < 4; ++j) acc[i][j] = vzero8();

  const int lrow = lane & 15;
  const int ac0  = (lane >> 4) * 8;
  const int bk0  = (lane >> 4) * 8;

  for (int k0 = 0; k0 < D_; k0 += 64) {
    // stage A (x) and B (W) 128x64 tiles as bf16
#pragma unroll
    for (int rep = 0; rep < 8; ++rep) {
      int idx = rep * 256 + tid;  // 0..2047
      int row = idx >> 4;         // 0..127
      int col = (idx & 15) * 4;   // 0..60
      float4 fa = *(const float4*)&x[((size_t)b * L_ + l0 + row) * D_ + k0 + col];
      As[row][col + 0] = (__bf16)fa.x;
      As[row][col + 1] = (__bf16)fa.y;
      As[row][col + 2] = (__bf16)fa.z;
      As[row][col + 3] = (__bf16)fa.w;
      float4 fb = *(const float4*)&W[(size_t)(n0 + row) * D_ + k0 + col];
      Bs[row][col + 0] = (__bf16)fb.x;
      Bs[row][col + 1] = (__bf16)fb.y;
      Bs[row][col + 2] = (__bf16)fb.z;
      Bs[row][col + 3] = (__bf16)fb.w;
    }
    __syncthreads();

#pragma unroll
    for (int ks = 0; ks < 2; ++ks) {  // two 32-deep WMMA K-steps per stage
      v16bf afr[2], bfr[4];
#pragma unroll
      for (int i = 0; i < 2; ++i)
        afr[i] = load_afrag(&As[wm + 16 * i + lrow][0], ks * 32 + ac0);
#pragma unroll
      for (int j = 0; j < 4; ++j)
        bfr[j] = load_bfrag(&Bs[wn + 16 * j + lrow][0], ks * 32 + bk0);
#pragma unroll
      for (int i = 0; i < 2; ++i)
#pragma unroll
        for (int j = 0; j < 4; ++j)
          acc[i][j] = wmma_bf16(afr[i], bfr[j], acc[i][j]);
    }
    __syncthreads();
  }

  // Store as bf16 into [B,H,L,HD]; head h = n0/128.
  const int h  = n0 >> 7;
  const int mo = 8 * (lane >> 4);
  const int nc = lane & 15;
#pragma unroll
  for (int i = 0; i < 2; ++i)
#pragma unroll
    for (int j = 0; j < 4; ++j)
#pragma unroll
      for (int r = 0; r < 8; ++r) {
        int m  = wm + 16 * i + r + mo;
        int hd = wn + 16 * j + nc;
        out[(((size_t)b * H_ + h) * L_ + l0 + m) * HD_ + hd] =
            (__bf16)(acc[i][j][r] * scale);
      }
}

// ---------------------------------------------------------------------------
// Kernel 2: retention attention for one (b,h) and one 64-row query tile.
// out[i,:] = sum_{j<=i} gamma^(i-j) (q_i . k_j) v_j   (scale folded into Q)
// ---------------------------------------------------------------------------
__global__ __launch_bounds__(256) void ret_attn(
    const __bf16* __restrict__ Qb, const __bf16* __restrict__ Kb,
    const __bf16* __restrict__ Vb, const float* __restrict__ gamma_logit,
    float* __restrict__ Obuf) {
  __shared__ __bf16 Qs[64][136];   // [i][hd], 272B stride
  __shared__ __bf16 Ks[64][136];   // [j][hd]
  __shared__ __bf16 Vst[128][72];  // [hd][j] (transposed), 144B stride
  __shared__ __bf16 Ss[64][72];    // [i][j] decayed scores (bf16)

  const int tid  = threadIdx.x;
  const int wid  = tid >> 5;
  const int lane = tid & 31;
  const int bh = blockIdx.y;
  const int h  = bh % H_;
  const int i0 = blockIdx.x * 64;

  const float gl    = gamma_logit[h];
  const float gamma = 1.0f / (1.0f + expf(-gl));
  const float log2g = log2f(gamma);

  const int lrow = lane & 15;
  const int hi   = lane >> 4;

  // load Q tile (64x128)
  const size_t qbase = ((size_t)bh * L_ + i0) * HD_;
#pragma unroll
  for (int rep = 0; rep < 4; ++rep) {
    int idx = rep * 256 + tid;
    int row = idx >> 4;         // 0..63
    int c8  = (idx & 15) * 8;   // 0..120
    *(v8bf*)&Qs[row][c8] = *(const v8bf*)&Qb[qbase + (size_t)row * HD_ + c8];
  }

  const int ow_m = wid & 3;   // 16 output rows
  const int ow_n = wid >> 2;  // 64 output cols
  v8f acc[4];
#pragma unroll
  for (int j = 0; j < 4; ++j) acc[j] = vzero8();

  for (int j0 = 0; j0 <= i0; j0 += 64) {
    __syncthreads();  // previous-iter LDS readers done (and Q loaded, iter 0)
    const size_t kbase = ((size_t)bh * L_ + j0) * HD_;
#pragma unroll
    for (int rep = 0; rep < 4; ++rep) {
      int idx = rep * 256 + tid;
      int row = idx >> 4;
      int c8  = (idx & 15) * 8;
      *(v8bf*)&Ks[row][c8] = *(const v8bf*)&Kb[kbase + (size_t)row * HD_ + c8];
      v8bf vv = *(const v8bf*)&Vb[kbase + (size_t)row * HD_ + c8];
#pragma unroll
      for (int e = 0; e < 8; ++e) Vst[c8 + e][row] = vv[e];
    }
    __syncthreads();

    // S = Q K^T  (64x64, K-dim = 128)
    const int sm = (wid & 3) * 16;
    const int sn = (wid >> 2) * 32;
    v8f sacc[2];
    sacc[0] = vzero8();
    sacc[1] = vzero8();
#pragma unroll
    for (int hd0 = 0; hd0 < 128; hd0 += 32) {
      v16bf aq = load_afrag(&Qs[sm + lrow][0], hd0 + hi * 8);
#pragma unroll
      for (int sub = 0; sub < 2; ++sub) {
        v16bf bk = load_bfrag(&Ks[sn + 16 * sub + lrow][0], hd0 + hi * 8);
        sacc[sub] = wmma_bf16(aq, bk, sacc[sub]);
      }
    }
    // decay mask + requantize to bf16
#pragma unroll
    for (int sub = 0; sub < 2; ++sub)
#pragma unroll
      for (int r = 0; r < 8; ++r) {
        int il = sm + r + 8 * hi;
        int jl = sn + 16 * sub + lrow;
        int dist = (i0 + il) - (j0 + jl);
        float w = (dist < 0) ? 0.0f
                             : sacc[sub][r] * exp2f(log2g * (float)dist);
        Ss[il][jl] = (__bf16)w;
      }
    __syncthreads();

    // out += S @ V  (K-dim = 64), B from transposed Vst
#pragma unroll
    for (int kk = 0; kk < 64; kk += 32) {
      v16bf as = load_afrag(&Ss[16 * ow_m + lrow][0], kk + hi * 8);
#pragma unroll
      for (int j = 0; j < 4; ++j) {
        v16bf bv = load_bfrag(&Vst[64 * ow_n + 16 * j + lrow][0], kk + hi * 8);
        acc[j] = wmma_bf16(as, bv, acc[j]);
      }
    }
  }

  // store f32 to [B,H,L,HD]
#pragma unroll
  for (int j = 0; j < 4; ++j)
#pragma unroll
    for (int r = 0; r < 8; ++r) {
      int m  = 16 * ow_m + r + 8 * hi;
      int hd = 64 * ow_n + 16 * j + lrow;
      Obuf[((size_t)bh * L_ + i0 + m) * HD_ + hd] = acc[j][r];
    }
}

// ---------------------------------------------------------------------------
// Kernel 3: GroupNorm per (b,h) over (L,HD) + affine, write [B,L,D] fp32.
// ---------------------------------------------------------------------------
__global__ __launch_bounds__(256) void ret_groupnorm(
    const float* __restrict__ Obuf, const float* __restrict__ gw,
    const float* __restrict__ gb, float* __restrict__ out) {
  __shared__ float red[256];
  __shared__ float red2[256];
  const int tid = threadIdx.x;
  const int bh  = blockIdx.x;
  const int b   = bh / H_;
  const int h   = bh % H_;
  const size_t base = (size_t)bh * L_ * HD_;
  const int N = L_ * HD_;

  float s = 0.0f, s2 = 0.0f;
  for (int i = tid * 4; i < N; i += 256 * 4) {
    float4 v = *(const float4*)&Obuf[base + i];
    s  += v.x + v.y + v.z + v.w;
    s2 += v.x * v.x + v.y * v.y + v.z * v.z + v.w * v.w;
  }
  red[tid] = s;
  red2[tid] = s2;
  __syncthreads();
  for (int off = 128; off > 0; off >>= 1) {
    if (tid < off) {
      red[tid] += red[tid + off];
      red2[tid] += red2[tid + off];
    }
    __syncthreads();
  }
  const float mean = red[0] / (float)N;
  const float var  = red2[0] / (float)N - mean * mean;
  const float rstd = rsqrtf(var + EPS_);

  for (int i = tid * 4; i < N; i += 256 * 4) {
    float4 v = *(const float4*)&Obuf[base + i];
    int l  = i >> 7;    // / HD_
    int hd = i & 127;   // % HD_
    float4 w   = *(const float4*)&gw[h * HD_ + hd];
    float4 bia = *(const float4*)&gb[h * HD_ + hd];
    float4 y;
    y.x = (v.x - mean) * rstd * w.x + bia.x;
    y.y = (v.y - mean) * rstd * w.y + bia.y;
    y.z = (v.z - mean) * rstd * w.z + bia.z;
    y.w = (v.w - mean) * rstd * w.w + bia.w;
    *(float4*)&out[((size_t)b * L_ + l) * D_ + h * HD_ + hd] = y;
  }
}

extern "C" void kernel_launch(void* const* d_in, const int* in_sizes, int n_in,
                              void* d_out, int out_size, void* d_ws,
                              size_t ws_size, hipStream_t stream) {
  const float* x  = (const float*)d_in[0];
  const float* Wq = (const float*)d_in[1];
  const float* Wk = (const float*)d_in[2];
  const float* Wv = (const float*)d_in[3];
  const float* gl = (const float*)d_in[4];
  const float* gw = (const float*)d_in[5];
  const float* gb = (const float*)d_in[6];
  float* out = (float*)d_out;

  const size_t n = (size_t)B_ * L_ * D_;
  __bf16* Qb = (__bf16*)d_ws;
  __bf16* Kb = Qb + n;
  __bf16* Vb = Kb + n;
  float*  Obuf = (float*)(Vb + n);

  dim3 gproj(L_ / 128, D_ / 128, B_);
  ret_proj_gemm<<<gproj, 256, 0, stream>>>(x, Wq, Qb, SCALE_Q);
  ret_proj_gemm<<<gproj, 256, 0, stream>>>(x, Wk, Kb, 1.0f);
  ret_proj_gemm<<<gproj, 256, 0, stream>>>(x, Wv, Vb, 1.0f);

  dim3 gattn(L_ / 64, B_ * H_);
  ret_attn<<<gattn, 256, 0, stream>>>(Qb, Kb, Vb, gl, Obuf);

  ret_groupnorm<<<dim3(B_ * H_), 256, 0, stream>>>(Obuf, gw, gb, out);
}
